// BDH_80513456931224
// MI455X (gfx1250) — matrix operand order repaired and checked by reading, not compiled, not run on hardware
//
#include <hip/hip_runtime.h>
#include <hip/hip_bf16.h>
#include <stdint.h>

#define B_  2
#define T_  2048
#define D_  256
#define NH_ 4
#define N_  4096
#define V_  256

typedef __attribute__((ext_vector_type(16))) __bf16 v16bf;
typedef __attribute__((ext_vector_type(8)))  float  v8f;

__device__ __forceinline__ __bf16 to_bf16(float f) {
    union { float f; uint32_t u; } v; v.f = f;
    uint32_t r = v.u + 0x7FFFu + ((v.u >> 16) & 1u);   // round-to-nearest-even
    union { uint16_t u; __bf16 b; } o; o.u = (uint16_t)(r >> 16);
    return o.b;
}
__device__ __forceinline__ float from_bf16(__bf16 b) {
    union { uint16_t u; __bf16 b; } i; i.b = b;
    union { uint32_t u; float f; } o; o.u = ((uint32_t)i.u) << 16;
    return o.f;
}

__device__ __forceinline__ v8f wmma_bf16(v16bf a, v16bf b, v8f c) {
    return __builtin_amdgcn_wmma_f32_16x16x32_bf16(false, a, false, b, (short)0, c,
                                                   false, false);
}

// A-fragment: 16x32 bf16, row-major A[M,K]. ISA layout: lanes 0-15 row m0+lane,
// K = {0..7,16..23}; lanes 16-31 row m0+lane-16, K = {8..15,24..31}.
__device__ __forceinline__ v16bf load_a(const __bf16* A, int lda, int m0, int k0, int lane) {
    int r  = m0 + (lane & 15);
    int kb = k0 + ((lane >> 4) << 3);
    const uint32_t* p = (const uint32_t*)(A + (size_t)r * lda + kb);
    union { uint32_t u[8]; v16bf v; } o;
#pragma unroll
    for (int i = 0; i < 4; ++i) o.u[i] = p[i];
#pragma unroll
    for (int i = 0; i < 4; ++i) o.u[4 + i] = p[8 + i];
    return o.v;
}

// B-fragment: 32x16 bf16 with B given TRANSPOSED (Bt row-major [N,K], Bt[c,k]=B[k,c]).
// ISA layout: VGPR v<4, lanes 0-15 hold rows k0+2v,k0+2v+1 at col=lane; lanes 16-31 rows k0+8+2v; v>=4 adds 16.
// With NT layout each lane reads K = {0..7,16..23} (lanes 0-15) / {8..15,24..31} (lanes 16-31) from one Bt row.
__device__ __forceinline__ v16bf load_b_nt(const __bf16* Bt, int ldb, int n0, int k0, int lane) {
    int c  = n0 + (lane & 15);
    int kb = k0 + ((lane >> 4) << 3);
    const uint32_t* p = (const uint32_t*)(Bt + (size_t)c * ldb + kb);
    union { uint32_t u[8]; v16bf v; } o;
#pragma unroll
    for (int i = 0; i < 4; ++i) o.u[i] = p[i];
#pragma unroll
    for (int i = 0; i < 4; ++i) o.u[4 + i] = p[8 + i];
    return o.v;
}

// C element (row,col) for accumulator vgpr j: row = m0 + j + 8*(lane>=16), col = n0 + (lane&15)
#define C_ROW(m0, j, lane) ((m0) + (j) + (((lane) >> 4) << 3))
#define C_COL(n0, lane)    ((n0) + ((lane) & 15))

__device__ __forceinline__ float wave_sum(float v) {
#pragma unroll
    for (int o = 16; o > 0; o >>= 1) v += __shfl_xor(v, o, 32);
    return v;
}

// ---------------------------------------------------------------------------
// Weight prep: transpose fp32 [G,R,C] -> bf16 [G,C,R]
__global__ void k_tcvt(const float* __restrict__ in, __bf16* __restrict__ out,
                       int G, int R, int C) {
    long total = (long)G * R * C;
    for (long o = (long)blockIdx.x * blockDim.x + threadIdx.x; o < total;
         o += (long)gridDim.x * blockDim.x) {
        long g = o / ((long)R * C);
        long rem = o - g * (long)R * C;
        long c = rem / R;
        long r = rem - c * R;
        out[o] = to_bf16(in[(g * R + r) * C + c]);
    }
}

// x = ln(embed[idx]); emit fp32 x, bf16 x, bf16 x^T [b,d,t]
__global__ __launch_bounds__(256) void k_embed_ln(const int* __restrict__ idx,
        const float* __restrict__ embed, float* __restrict__ x,
        __bf16* __restrict__ xb, __bf16* __restrict__ xT) {
    int row  = blockIdx.x * 8 + (threadIdx.x >> 5);
    int lane = threadIdx.x & 31;
    int tok  = idx[row];
    const float* e = embed + (size_t)tok * D_;
    float f[8], s = 0.f;
#pragma unroll
    for (int i = 0; i < 8; ++i) { f[i] = e[lane + 32 * i]; s += f[i]; }
    float mean = wave_sum(s) * (1.f / D_);
    float vs = 0.f;
#pragma unroll
    for (int i = 0; i < 8; ++i) { f[i] -= mean; vs += f[i] * f[i]; }
    float rstd = rsqrtf(wave_sum(vs) * (1.f / D_) + 1e-5f);
    int b = row >> 11, t = row & (T_ - 1);
#pragma unroll
    for (int i = 0; i < 8; ++i) {
        int d = lane + 32 * i;
        float y = f[i] * rstd;
        x[(size_t)row * D_ + d]  = y;
        xb[(size_t)row * D_ + d] = to_bf16(y);
        xT[((size_t)b * D_ + d) * T_ + t] = to_bf16(y);
    }
}

// LayerNorm rows of fp32 [rows, 256] -> bf16 same layout
__global__ __launch_bounds__(256) void k_ln_rows(const float* __restrict__ in,
                                                 __bf16* __restrict__ out) {
    int row  = blockIdx.x * 8 + (threadIdx.x >> 5);
    int lane = threadIdx.x & 31;
    const float* p = in + (size_t)row * D_;
    float f[8], s = 0.f;
#pragma unroll
    for (int i = 0; i < 8; ++i) { f[i] = p[lane + 32 * i]; s += f[i]; }
    float mean = wave_sum(s) * (1.f / D_);
    float vs = 0.f;
#pragma unroll
    for (int i = 0; i < 8; ++i) { f[i] -= mean; vs += f[i] * f[i]; }
    float rstd = rsqrtf(wave_sum(vs) * (1.f / D_) + 1e-5f);
#pragma unroll
    for (int i = 0; i < 8; ++i)
        out[(size_t)row * D_ + lane + 32 * i] = to_bf16(f[i] * rstd);
}

// x = ln(x + ln(yMLP)); refresh fp32 x, bf16 x, bf16 x^T
__global__ __launch_bounds__(256) void k_resid_ln(const float* __restrict__ ymlp,
        float* __restrict__ x, __bf16* __restrict__ xb, __bf16* __restrict__ xT) {
    int row  = blockIdx.x * 8 + (threadIdx.x >> 5);
    int lane = threadIdx.x & 31;
    const float* p = ymlp + (size_t)row * D_;
    float f[8], s = 0.f;
#pragma unroll
    for (int i = 0; i < 8; ++i) { f[i] = p[lane + 32 * i]; s += f[i]; }
    float mean = wave_sum(s) * (1.f / D_);
    float vs = 0.f;
#pragma unroll
    for (int i = 0; i < 8; ++i) { f[i] -= mean; vs += f[i] * f[i]; }
    float rstd = rsqrtf(wave_sum(vs) * (1.f / D_) + 1e-5f);
    // residual add then second LN
    float g[8]; s = 0.f;
#pragma unroll
    for (int i = 0; i < 8; ++i) {
        g[i] = x[(size_t)row * D_ + lane + 32 * i] + f[i] * rstd;
        s += g[i];
    }
    mean = wave_sum(s) * (1.f / D_);
    vs = 0.f;
#pragma unroll
    for (int i = 0; i < 8; ++i) { g[i] -= mean; vs += g[i] * g[i]; }
    rstd = rsqrtf(wave_sum(vs) * (1.f / D_) + 1e-5f);
    int b = row >> 11, t = row & (T_ - 1);
#pragma unroll
    for (int i = 0; i < 8; ++i) {
        int d = lane + 32 * i;
        float y = g[i] * rstd;
        x[(size_t)row * D_ + d]  = y;
        xb[(size_t)row * D_ + d] = to_bf16(y);
        xT[((size_t)b * D_ + d) * T_ + t] = to_bf16(y);
    }
}

// ---------------------------------------------------------------------------
// Encoder GEMM + relu + RoPE: x_latent[b,h] = x[b] (T x D) * encT[h]^T (D x N)
__global__ __launch_bounds__(128) void k_enc(const __bf16* __restrict__ xb,
        const __bf16* __restrict__ encT, __bf16* __restrict__ xsp,
        __bf16* __restrict__ qr) {
    int lane = threadIdx.x & 31, wave = threadIdx.x >> 5;
    int z = blockIdx.z, b = z >> 2, h = z & 3;
    int m0 = blockIdx.y * 64 + (wave >> 1) * 32;
    int n0 = blockIdx.x * 64 + (wave & 1) * 32;
    const __bf16* A  = xb + (size_t)b * T_ * D_;
    const __bf16* Bt = encT + (size_t)h * N_ * D_;
    v8f acc[2][2] = {};
    for (int k = 0; k < D_; k += 32) {
        v16bf a0 = load_a(A, D_, m0, k, lane);
        v16bf a1 = load_a(A, D_, m0 + 16, k, lane);
        v16bf b0 = load_b_nt(Bt, D_, n0, k, lane);
        v16bf b1 = load_b_nt(Bt, D_, n0 + 16, k, lane);
        acc[0][0] = wmma_bf16(a0, b0, acc[0][0]);
        acc[0][1] = wmma_bf16(a0, b1, acc[0][1]);
        acc[1][0] = wmma_bf16(a1, b0, acc[1][0]);
        acc[1][1] = wmma_bf16(a1, b1, acc[1][1]);
    }
    size_t base = (size_t)z * T_ * N_;
#pragma unroll
    for (int im = 0; im < 2; ++im)
#pragma unroll
    for (int in = 0; in < 2; ++in)
#pragma unroll
    for (int j = 0; j < 8; ++j) {
        int r = C_ROW(m0 + im * 16, j, lane);
        int c = C_COL(n0 + in * 16, lane);
        float sgn = fmaxf(acc[im][in][j], 0.f);          // relu -> x_sparse
        float partner = __shfl_xor(sgn, 1, 32);          // RoPE pair (n^1) is lane^1
        float rot = (c & 1) ? partner : -partner;
        float fr = exp2f((float)(c & ~1) * (-16.0f / (float)N_)) * 0.15915494309189535f;
        float ph = (float)r * fr;
        ph = (ph - floorf(ph)) * 6.283185307179586f;
        float sn, cs; __sincosf(ph, &sn, &cs);
        size_t o = base + (size_t)r * N_ + c;
        xsp[o] = to_bf16(sgn);
        qr[o]  = to_bf16(sgn * cs + rot * sn);
    }
}

// scores[b,h] = QR * QR^T, strictly-lower masked. Upper tiles: zero-fill, no K-loop.
__global__ __launch_bounds__(128) void k_scores(const __bf16* __restrict__ qr,
                                                __bf16* __restrict__ sc) {
    int lane = threadIdx.x & 31, wave = threadIdx.x >> 5;
    int z = blockIdx.z;
    int m0 = blockIdx.y * 64 + (wave >> 1) * 32;   // t
    int n0 = blockIdx.x * 64 + (wave & 1) * 32;    // s
    const __bf16* Q = qr + (size_t)z * T_ * N_;
    __bf16* S = sc + (size_t)z * T_ * T_;
    if (n0 > m0) {                                  // fully above diagonal: all masked
        __bf16 zb = to_bf16(0.f);
#pragma unroll
        for (int im = 0; im < 2; ++im)
#pragma unroll
        for (int in = 0; in < 2; ++in)
#pragma unroll
        for (int j = 0; j < 8; ++j)
            S[(size_t)C_ROW(m0 + im * 16, j, lane) * T_ + C_COL(n0 + in * 16, lane)] = zb;
        return;
    }
    v8f acc[2][2] = {};
    for (int k = 0; k < N_; k += 32) {
        v16bf a0 = load_a(Q, N_, m0, k, lane);
        v16bf a1 = load_a(Q, N_, m0 + 16, k, lane);
        v16bf b0 = load_b_nt(Q, N_, n0, k, lane);   // KR == QR, NT layout is natural
        v16bf b1 = load_b_nt(Q, N_, n0 + 16, k, lane);
        acc[0][0] = wmma_bf16(a0, b0, acc[0][0]);
        acc[0][1] = wmma_bf16(a0, b1, acc[0][1]);
        acc[1][0] = wmma_bf16(a1, b0, acc[1][0]);
        acc[1][1] = wmma_bf16(a1, b1, acc[1][1]);
    }
#pragma unroll
    for (int im = 0; im < 2; ++im)
#pragma unroll
    for (int in = 0; in < 2; ++in)
#pragma unroll
    for (int j = 0; j < 8; ++j) {
        int r = C_ROW(m0 + im * 16, j, lane);
        int c = C_COL(n0 + in * 16, lane);
        float v = (c < r) ? acc[im][in][j] : 0.f;   // strict lower mask
        S[(size_t)r * T_ + c] = to_bf16(v);
    }
}

// yKV[b,h] = scores[b,h] (T x T) * x[b] (T x D); K-loop truncated to k < t0+32
__global__ __launch_bounds__(128) void k_ykv(const __bf16* __restrict__ sc,
        const __bf16* __restrict__ xT, float* __restrict__ ykv) {
    int lane = threadIdx.x & 31, wave = threadIdx.x >> 5;
    int z = blockIdx.z, b = z >> 2;
    int m0 = blockIdx.y * 64 + (wave >> 1) * 32;
    int n0 = blockIdx.x * 64 + (wave & 1) * 32;
    const __bf16* A  = sc + (size_t)z * T_ * T_;
    const __bf16* Bt = xT + (size_t)b * D_ * T_;
    v8f acc[2][2] = {};
    int khi = m0 + 32;                               // scores[t,s]==0 for s>=t
    for (int k = 0; k < khi; k += 32) {
        v16bf a0 = load_a(A, T_, m0, k, lane);
        v16bf a1 = load_a(A, T_, m0 + 16, k, lane);
        v16bf b0 = load_b_nt(Bt, T_, n0, k, lane);
        v16bf b1 = load_b_nt(Bt, T_, n0 + 16, k, lane);
        acc[0][0] = wmma_bf16(a0, b0, acc[0][0]);
        acc[0][1] = wmma_bf16(a0, b1, acc[0][1]);
        acc[1][0] = wmma_bf16(a1, b0, acc[1][0]);
        acc[1][1] = wmma_bf16(a1, b1, acc[1][1]);
    }
    float* C = ykv + (size_t)z * T_ * D_;
#pragma unroll
    for (int im = 0; im < 2; ++im)
#pragma unroll
    for (int in = 0; in < 2; ++in)
#pragma unroll
    for (int j = 0; j < 8; ++j)
        C[(size_t)C_ROW(m0 + im * 16, j, lane) * D_ + C_COL(n0 + in * 16, lane)] =
            acc[im][in][j];
}

// y_sparse = relu(ln(yKV) * encoder_v), gated: xy = y_sparse * x_sparse
__global__ __launch_bounds__(128) void k_encv(const __bf16* __restrict__ yln,
        const __bf16* __restrict__ encvT, const __bf16* __restrict__ xsp,
        __bf16* __restrict__ xy) {
    int lane = threadIdx.x & 31, wave = threadIdx.x >> 5;
    int z = blockIdx.z, h = z & 3;
    int m0 = blockIdx.y * 64 + (wave >> 1) * 32;
    int n0 = blockIdx.x * 64 + (wave & 1) * 32;
    const __bf16* A  = yln + (size_t)z * T_ * D_;
    const __bf16* Bt = encvT + (size_t)h * N_ * D_;
    v8f acc[2][2] = {};
    for (int k = 0; k < D_; k += 32) {
        v16bf a0 = load_a(A, D_, m0, k, lane);
        v16bf a1 = load_a(A, D_, m0 + 16, k, lane);
        v16bf b0 = load_b_nt(Bt, D_, n0, k, lane);
        v16bf b1 = load_b_nt(Bt, D_, n0 + 16, k, lane);
        acc[0][0] = wmma_bf16(a0, b0, acc[0][0]);
        acc[0][1] = wmma_bf16(a0, b1, acc[0][1]);
        acc[1][0] = wmma_bf16(a1, b0, acc[1][0]);
        acc[1][1] = wmma_bf16(a1, b1, acc[1][1]);
    }
    size_t base = (size_t)z * T_ * N_;
#pragma unroll
    for (int im = 0; im < 2; ++im)
#pragma unroll
    for (int in = 0; in < 2; ++in)
#pragma unroll
    for (int j = 0; j < 8; ++j) {
        size_t o = base + (size_t)C_ROW(m0 + im * 16, j, lane) * N_
                        + C_COL(n0 + in * 16, lane);
        float y = fmaxf(acc[im][in][j], 0.f);
        xy[o] = to_bf16(y * from_bf16(xsp[o]));
    }
}

// yMLP[b] = sum_h xy[b,h] (T x N) * decT[h]^T (N x D)
__global__ __launch_bounds__(128) void k_dec(const __bf16* __restrict__ xy,
        const __bf16* __restrict__ decT, float* __restrict__ ymlp) {
    int lane = threadIdx.x & 31, wave = threadIdx.x >> 5;
    int b = blockIdx.z;
    int m0 = blockIdx.y * 64 + (wave >> 1) * 32;
    int n0 = blockIdx.x * 64 + (wave & 1) * 32;
    v8f acc[2][2] = {};
    for (int h = 0; h < NH_; ++h) {
        const __bf16* A  = xy + ((size_t)(b * NH_ + h)) * T_ * N_;
        const __bf16* Bt = decT + (size_t)h * D_ * N_;
        for (int k = 0; k < N_; k += 32) {
            v16bf a0 = load_a(A, N_, m0, k, lane);
            v16bf a1 = load_a(A, N_, m0 + 16, k, lane);
            v16bf b0 = load_b_nt(Bt, N_, n0, k, lane);
            v16bf b1 = load_b_nt(Bt, N_, n0 + 16, k, lane);
            acc[0][0] = wmma_bf16(a0, b0, acc[0][0]);
            acc[0][1] = wmma_bf16(a0, b1, acc[0][1]);
            acc[1][0] = wmma_bf16(a1, b0, acc[1][0]);
            acc[1][1] = wmma_bf16(a1, b1, acc[1][1]);
        }
    }
    float* C = ymlp + (size_t)b * T_ * D_;
#pragma unroll
    for (int im = 0; im < 2; ++im)
#pragma unroll
    for (int in = 0; in < 2; ++in)
#pragma unroll
    for (int j = 0; j < 8; ++j)
        C[(size_t)C_ROW(m0 + im * 16, j, lane) * D_ + C_COL(n0 + in * 16, lane)] =
            acc[im][in][j];
}

// logits = x (B*T x D) * lmT^T (D x V), fp32 out
__global__ __launch_bounds__(128) void k_lmhead(const __bf16* __restrict__ xb,
        const __bf16* __restrict__ lmT, float* __restrict__ out) {
    int lane = threadIdx.x & 31, wave = threadIdx.x >> 5;
    int m0 = blockIdx.y * 64 + (wave >> 1) * 32;
    int n0 = blockIdx.x * 64 + (wave & 1) * 32;
    v8f acc[2][2] = {};
    for (int k = 0; k < D_; k += 32) {
        v16bf a0 = load_a(xb, D_, m0, k, lane);
        v16bf a1 = load_a(xb, D_, m0 + 16, k, lane);
        v16bf b0 = load_b_nt(lmT, D_, n0, k, lane);
        v16bf b1 = load_b_nt(lmT, D_, n0 + 16, k, lane);
        acc[0][0] = wmma_bf16(a0, b0, acc[0][0]);
        acc[0][1] = wmma_bf16(a0, b1, acc[0][1]);
        acc[1][0] = wmma_bf16(a1, b0, acc[1][0]);
        acc[1][1] = wmma_bf16(a1, b1, acc[1][1]);
    }
#pragma unroll
    for (int im = 0; im < 2; ++im)
#pragma unroll
    for (int in = 0; in < 2; ++in)
#pragma unroll
    for (int j = 0; j < 8; ++j)
        out[(size_t)C_ROW(m0 + im * 16, j, lane) * V_ + C_COL(n0 + in * 16, lane)] =
            acc[im][in][j];
}

// ---------------------------------------------------------------------------
extern "C" void kernel_launch(void* const* d_in, const int* in_sizes, int n_in,
                              void* d_out, int out_size, void* d_ws, size_t ws_size,
                              hipStream_t stream) {
    (void)in_sizes; (void)n_in; (void)out_size; (void)ws_size;
    const int*   idx       = (const int*)d_in[0];
    const float* embed     = (const float*)d_in[1];
    const float* encoder   = (const float*)d_in[2];
    const float* encoder_v = (const float*)d_in[3];
    const float* decoder   = (const float*)d_in[4];
    const float* lm_head   = (const float*)d_in[5];
    float* out = (float*)d_out;

    char* w = (char*)d_ws;
    auto alloc = [&](size_t bytes) -> char* {
        char* p = w;
        w += (bytes + 255) & ~(size_t)255;
        return p;
    };
    __bf16* encT  = (__bf16*)alloc((size_t)NH_ * N_ * D_ * 2);      // [h,n,d]
    __bf16* encvT = (__bf16*)alloc((size_t)NH_ * N_ * D_ * 2);      // [h,n,d]
    __bf16* decT  = (__bf16*)alloc((size_t)NH_ * D_ * N_ * 2);      // [h,d,n]
    __bf16* lmT   = (__bf16*)alloc((size_t)V_ * D_ * 2);            // [v,d]
    float*  x     = (float*)alloc((size_t)B_ * T_ * D_ * 4);
    __bf16* xb    = (__bf16*)alloc((size_t)B_ * T_ * D_ * 2);
    __bf16* xT    = (__bf16*)alloc((size_t)B_ * D_ * T_ * 2);       // [b,d,t]
    __bf16* xsp   = (__bf16*)alloc((size_t)B_ * NH_ * T_ * N_ * 2); // x_sparse
    __bf16* qr    = (__bf16*)alloc((size_t)B_ * NH_ * T_ * N_ * 2); // QR, reused as xy
    __bf16* sc    = (__bf16*)alloc((size_t)B_ * NH_ * T_ * T_ * 2); // masked scores
    float*  ykv   = (float*)alloc((size_t)B_ * NH_ * T_ * D_ * 4);
    __bf16* yln   = (__bf16*)alloc((size_t)B_ * NH_ * T_ * D_ * 2);
    float*  ymlp  = (float*)alloc((size_t)B_ * T_ * D_ * 4);

    // one-time weight transpose + bf16 convert
    k_tcvt<<<4096, 256, 0, stream>>>(encoder,   encT,  NH_, D_, N_);
    k_tcvt<<<4096, 256, 0, stream>>>(encoder_v, encvT, NH_, D_, N_);
    k_tcvt<<<4096, 256, 0, stream>>>(decoder,   decT,  NH_, N_, D_);
    k_tcvt<<<256,  256, 0, stream>>>(lm_head,   lmT,   1,   D_, V_);

    k_embed_ln<<<(B_ * T_) / 8, 256, 0, stream>>>(idx, embed, x, xb, xT);

    for (int layer = 0; layer < 6; ++layer) {
        k_enc<<<dim3(N_ / 64, T_ / 64, B_ * NH_), 128, 0, stream>>>(xb, encT, xsp, qr);
        k_scores<<<dim3(T_ / 64, T_ / 64, B_ * NH_), 128, 0, stream>>>(qr, sc);
        k_ykv<<<dim3(D_ / 64, T_ / 64, B_ * NH_), 128, 0, stream>>>(sc, xT, ykv);
        k_ln_rows<<<(B_ * NH_ * T_) / 8, 256, 0, stream>>>(ykv, yln);
        k_encv<<<dim3(N_ / 64, T_ / 64, B_ * NH_), 128, 0, stream>>>(yln, encvT, xsp,
                                                                     qr /*xy*/);
        k_dec<<<dim3(D_ / 64, T_ / 64, B_), 128, 0, stream>>>(qr /*xy*/, decT, ymlp);
        k_resid_ln<<<(B_ * T_) / 8, 256, 0, stream>>>(ymlp, x, xb, xT);
    }
    k_lmhead<<<dim3(V_ / 64, (B_ * T_) / 64, 1), 128, 0, stream>>>(xb, lmT, out);
}
